// EnrichBlock_9388798509359
// MI455X (gfx1250) — matrix operand order repaired and checked by reading, not compiled, not run on hardware
//
#include <hip/hip_runtime.h>
#include <hip/hip_bf16.h>
#include <stdint.h>

// ---------------------------------------------------------------------------
// Types for CDNA5 WMMA
// ---------------------------------------------------------------------------
typedef __bf16 bf16;
typedef __attribute__((ext_vector_type(16))) __bf16 v16bf;
typedef __attribute__((ext_vector_type(8)))  float  v8f;

#define DV  512
#define FFV 2048
#define SV  2048
#define BV  2
#define HV  8
#define BSV (BV * SV)   // 4096 rows

// ---------------------------------------------------------------------------
// CDNA5 async global->LDS copy (ASYNCcnt path, 16B per lane)
// ---------------------------------------------------------------------------
__device__ __forceinline__ void async_ld_b128(const bf16* gsrc, bf16* ldst) {
  unsigned       l = (unsigned)(uintptr_t)ldst;          // low 32b = LDS offset
  unsigned long long g = (unsigned long long)(uintptr_t)gsrc;
  asm volatile("global_load_async_to_lds_b128 %0, %1, off"
               :: "v"(l), "v"(g) : "memory");
}
__device__ __forceinline__ void wait_async0() {
  asm volatile("s_wait_asynccnt 0" ::: "memory");
}

// ---------------------------------------------------------------------------
// fp32 -> bf16 conversion (vectorized x4)
// ---------------------------------------------------------------------------
__global__ __launch_bounds__(256) void cvt_bf16_kernel(const float* __restrict__ in,
                                                       bf16* __restrict__ out, int n) {
  int i = (blockIdx.x * 256 + threadIdx.x) * 4;
  if (i < n) {
    float4 v = *(const float4*)(in + i);
    out[i + 0] = (bf16)v.x;
    out[i + 1] = (bf16)v.y;
    out[i + 2] = (bf16)v.z;
    out[i + 3] = (bf16)v.w;
  }
}

// ---------------------------------------------------------------------------
// WMMA GEMM:  C[M][N] = A[M][K] * W[N][K]^T + bias[N]   (optionally ReLU)
// Block tile 128x128, 256 thr = 8 waves (4Mx2N), wave 32x64 = 2x4 WMMA tiles.
// Double-buffered LDS with async global->LDS staging.
// ---------------------------------------------------------------------------
__global__ __launch_bounds__(256, 2) void gemm_wmma_kernel(
    const bf16* __restrict__ A, const bf16* __restrict__ W,
    const float* __restrict__ bias,
    float* __restrict__ outF, bf16* __restrict__ outB,
    int M, int N, int K, int relu) {
  __shared__ bf16 As[2][128][40];  // 32 + 8 pad (80B row stride, 16B aligned)
  __shared__ bf16 Bs[2][128][40];

  const int bm = blockIdx.x * 128;
  const int bn = blockIdx.y * 128;
  const int tid  = threadIdx.x;
  const int lane = tid & 31;
  const int wave = tid >> 5;
  const int wm = (wave >> 1) * 32;  // 0,32,64,96
  const int wn = (wave & 1) * 64;   // 0,64
  const int half = lane >> 4;
  const int l15  = lane & 15;

  v8f acc[2][4];
#pragma unroll
  for (int mi = 0; mi < 2; ++mi)
#pragma unroll
    for (int ni = 0; ni < 4; ++ni)
#pragma unroll
      for (int r = 0; r < 8; ++r) acc[mi][ni][r] = 0.0f;

  // async stage of one 128x32 A tile + 128x32 B tile (2+2 b128 per thread)
  auto stage = [&](int buf, int k0) {
    const int row = tid >> 1;
    const int seg = (tid & 1) * 16;
    async_ld_b128(A + (size_t)(bm + row) * K + k0 + seg,     &As[buf][row][seg]);
    async_ld_b128(A + (size_t)(bm + row) * K + k0 + seg + 8, &As[buf][row][seg + 8]);
    async_ld_b128(W + (size_t)(bn + row) * K + k0 + seg,     &Bs[buf][row][seg]);
    async_ld_b128(W + (size_t)(bn + row) * K + k0 + seg + 8, &Bs[buf][row][seg + 8]);
  };

  stage(0, 0);
  wait_async0();
  __syncthreads();

  int buf = 0;
  for (int k0 = 0; k0 < K; k0 += 32) {
    if (k0 + 32 < K) stage(buf ^ 1, k0 + 32);  // prefetch next tile (async)

    // A fragments: 16x32 layout. lane<16: row=l15, K={half*8+e, 16+half*8+e}
    v16bf afr[2];
#pragma unroll
    for (int mi = 0; mi < 2; ++mi) {
      const int arow = wm + mi * 16 + l15;
#pragma unroll
      for (int e = 0; e < 8; ++e) {
        afr[mi][e]     = As[buf][arow][half * 8 + e];
        afr[mi][8 + e] = As[buf][arow][16 + half * 8 + e];
      }
    }
    // B fragments: 32x16 layout. lane: n=l15, K={half*8+e, 16+half*8+e}
    v16bf bfr[4];
#pragma unroll
    for (int ni = 0; ni < 4; ++ni) {
      const int brow = wn + ni * 16 + l15;
#pragma unroll
      for (int e = 0; e < 8; ++e) {
        bfr[ni][e]     = Bs[buf][brow][half * 8 + e];
        bfr[ni][8 + e] = Bs[buf][brow][16 + half * 8 + e];
      }
    }
#pragma unroll
    for (int mi = 0; mi < 2; ++mi)
#pragma unroll
      for (int ni = 0; ni < 4; ++ni)
        acc[mi][ni] = __builtin_amdgcn_wmma_f32_16x16x32_bf16(
            false, afr[mi], false, bfr[ni], (short)0, acc[mi][ni], false, false);

    wait_async0();
    __syncthreads();
    buf ^= 1;
  }

  // Epilogue: C layout — VGPR r, lane: row = r + 8*half, col = l15
#pragma unroll
  for (int mi = 0; mi < 2; ++mi) {
#pragma unroll
    for (int ni = 0; ni < 4; ++ni) {
      const int colAbs = bn + wn + ni * 16 + l15;
      const float bv = bias[colAbs];
#pragma unroll
      for (int r = 0; r < 8; ++r) {
        const int rowAbs = bm + wm + mi * 16 + r + 8 * half;
        float v = acc[mi][ni][r] + bv;
        if (relu) v = fmaxf(v, 0.0f);
        const size_t idx = (size_t)rowAbs * N + colAbs;
        if (outF) outF[idx] = v;
        if (outB) outB[idx] = (bf16)v;
      }
    }
  }
}

// ---------------------------------------------------------------------------
// Flash attention (per head): Q,K,V bf16 [B*S][512] with head at col h*64.
// Block = 8 waves; each wave owns a 16-row Q tile; double-buffered async
// 32-key K/V tiles. scores = QK^T/8 (+causal), online softmax, O += P V.
// ---------------------------------------------------------------------------
__global__ __launch_bounds__(256, 2) void attn_wmma_kernel(
    const bf16* __restrict__ Q, const bf16* __restrict__ Kg,
    const bf16* __restrict__ Vg, float* __restrict__ outF,
    bf16* __restrict__ outB, int S, int causal) {
  __shared__ bf16 Ks[2][32][72];
  __shared__ bf16 Vs[2][32][72];
  __shared__ bf16 Ps[8][16][40];

  const int nqt = S / 128;
  const int t  = blockIdx.x;
  const int qt = t % nqt;
  const int h  = (t / nqt) % HV;
  const int b  = t / (nqt * HV);
  const int q0b = qt * 128;
  const int tid  = threadIdx.x;
  const int lane = tid & 31;
  const int wave = tid >> 5;
  const int half = lane >> 4;
  const int l15  = lane & 15;
  const int hc   = h * 64;
  const size_t rowBase = (size_t)b * S;
  const int q0 = q0b + wave * 16;

  // Q fragments (A layout, two K-chunks of 32 over DK=64)
  v16bf aq[2];
#pragma unroll
  for (int j = 0; j < 2; ++j) {
    const bf16* qp = Q + (rowBase + q0 + l15) * DV + hc + j * 32 + half * 8;
    union { uint4 u[2]; bf16 h16[16]; } uu;
    uu.u[0] = *(const uint4*)qp;
    uu.u[1] = *(const uint4*)(qp + 16);
#pragma unroll
    for (int e = 0; e < 16; ++e) aq[j][e] = uu.h16[e];
  }

  v8f oacc[4];
  float mI[8], lI[8];
#pragma unroll
  for (int di = 0; di < 4; ++di)
#pragma unroll
    for (int r = 0; r < 8; ++r) oacc[di][r] = 0.0f;
#pragma unroll
  for (int r = 0; r < 8; ++r) { mI[r] = -1e30f; lI[r] = 0.0f; }

  const int kend  = causal ? (q0b + 128) : S;
  const int myend = causal ? (q0 + 16) : S;

  auto stageKV = [&](int buf, int kc) {
    const int row = tid >> 3;
    const int seg = (tid & 7) * 8;
    async_ld_b128(Kg + (rowBase + kc + row) * DV + hc + seg, &Ks[buf][row][seg]);
    async_ld_b128(Vg + (rowBase + kc + row) * DV + hc + seg, &Vs[buf][row][seg]);
  };

  stageKV(0, 0);
  wait_async0();
  __syncthreads();

  int buf = 0;
  for (int kc = 0; kc < kend; kc += 32) {
    if (kc + 32 < kend) stageKV(buf ^ 1, kc + 32);  // async prefetch next keys

    if (kc < myend) {
      // scores: 2 N-tiles x 2 K-chunks of WMMA
      v8f sc[2];
#pragma unroll
      for (int ni = 0; ni < 2; ++ni)
#pragma unroll
        for (int r = 0; r < 8; ++r) sc[ni][r] = 0.0f;
#pragma unroll
      for (int j = 0; j < 2; ++j) {
#pragma unroll
        for (int ni = 0; ni < 2; ++ni) {
          v16bf bk;  // B[dk][key]: lane n = key, K = j*32 + {half*8+e, 16+half*8+e}
#pragma unroll
          for (int e = 0; e < 8; ++e) {
            bk[e]     = Ks[buf][ni * 16 + l15][j * 32 + half * 8 + e];
            bk[8 + e] = Ks[buf][ni * 16 + l15][j * 32 + 16 + half * 8 + e];
          }
          sc[ni] = __builtin_amdgcn_wmma_f32_16x16x32_bf16(
              false, aq[j], false, bk, (short)0, sc[ni], false, false);
        }
      }
      // scale + causal mask + online softmax (row stats in 16-lane halves)
      float p[2][8];
#pragma unroll
      for (int r = 0; r < 8; ++r) {
        const int rowAbs = q0 + r + 8 * half;
#pragma unroll
        for (int ni = 0; ni < 2; ++ni) {
          float s = sc[ni][r] * 0.125f;  // 1/sqrt(64)
          const int col = kc + ni * 16 + l15;
          if (causal && col > rowAbs) s = -1e9f;
          p[ni][r] = s;
        }
        float v = fmaxf(p[0][r], p[1][r]);
        v = fmaxf(v, __shfl_xor(v, 1, 32));
        v = fmaxf(v, __shfl_xor(v, 2, 32));
        v = fmaxf(v, __shfl_xor(v, 4, 32));
        v = fmaxf(v, __shfl_xor(v, 8, 32));
        const float mnew  = fmaxf(mI[r], v);
        const float alpha = __expf(mI[r] - mnew);
        float rs = 0.0f;
#pragma unroll
        for (int ni = 0; ni < 2; ++ni) {
          const float e = __expf(p[ni][r] - mnew);
          p[ni][r] = e;
          rs += e;
        }
        rs += __shfl_xor(rs, 1, 32);
        rs += __shfl_xor(rs, 2, 32);
        rs += __shfl_xor(rs, 4, 32);
        rs += __shfl_xor(rs, 8, 32);
        lI[r] = lI[r] * alpha + rs;
        mI[r] = mnew;
#pragma unroll
        for (int di = 0; di < 4; ++di) oacc[di][r] *= alpha;
        // park P (C layout) into LDS as bf16 for A-layout reload
#pragma unroll
        for (int ni = 0; ni < 2; ++ni)
          Ps[wave][r + 8 * half][ni * 16 + l15] = (bf16)p[ni][r];
      }
      asm volatile("s_wait_dscnt 0" ::: "memory");  // wave-local LDS RAW fence
      v16bf pa;  // P as A-matrix 16x32
#pragma unroll
      for (int e = 0; e < 8; ++e) {
        pa[e]     = Ps[wave][l15][half * 8 + e];
        pa[8 + e] = Ps[wave][l15][16 + half * 8 + e];
      }
      // O += P(16x32) * V(32x64)
#pragma unroll
      for (int di = 0; di < 4; ++di) {
        v16bf bv;  // B[k][d]: lane n = d, K = {half*8+e, 16+half*8+e}
#pragma unroll
        for (int e = 0; e < 8; ++e) {
          bv[e]     = Vs[buf][half * 8 + e][di * 16 + l15];
          bv[8 + e] = Vs[buf][16 + half * 8 + e][di * 16 + l15];
        }
        oacc[di] = __builtin_amdgcn_wmma_f32_16x16x32_bf16(
            false, pa, false, bv, (short)0, oacc[di], false, false);
      }
    }
    wait_async0();
    __syncthreads();
    buf ^= 1;
  }

  // normalize + store (merged-heads layout [B*S][512])
#pragma unroll
  for (int r = 0; r < 8; ++r) {
    const float inv = 1.0f / lI[r];
    const int rowAbs = q0 + r + 8 * half;
#pragma unroll
    for (int di = 0; di < 4; ++di) {
      const float o = oacc[di][r] * inv;
      const size_t idx = (rowBase + rowAbs) * DV + hc + di * 16 + l15;
      if (outF) outF[idx] = o;
      if (outB) outB[idx] = (bf16)o;
    }
  }
}

// ---------------------------------------------------------------------------
// LayerNorm(x + res) * g + b  -> fp32 stream + bf16 mirror. 1 row / block.
// ---------------------------------------------------------------------------
__global__ __launch_bounds__(128) void ln_kernel(
    const float* __restrict__ x, const float* __restrict__ res,
    const float* __restrict__ g, const float* __restrict__ bt,
    float* __restrict__ outF, bf16* __restrict__ outB) {
  __shared__ float red[4];
  const int row  = blockIdx.x;
  const int tid  = threadIdx.x;
  const int lane = tid & 31;
  const int wave = tid >> 5;

  float4 v = ((const float4*)(x + (size_t)row * DV))[tid];
  if (res) {
    float4 r = ((const float4*)(res + (size_t)row * DV))[tid];
    v.x += r.x; v.y += r.y; v.z += r.z; v.w += r.w;
  }
  float s = v.x + v.y + v.z + v.w;
  for (int m = 1; m < 32; m <<= 1) s += __shfl_xor(s, m, 32);
  if (lane == 0) red[wave] = s;
  __syncthreads();
  const float mean = (red[0] + red[1] + red[2] + red[3]) * (1.0f / 512.0f);
  __syncthreads();

  const float dx = v.x - mean, dy = v.y - mean, dz = v.z - mean, dw = v.w - mean;
  float q = dx * dx + dy * dy + dz * dz + dw * dw;
  for (int m = 1; m < 32; m <<= 1) q += __shfl_xor(q, m, 32);
  if (lane == 0) red[wave] = q;
  __syncthreads();
  const float var  = (red[0] + red[1] + red[2] + red[3]) * (1.0f / 512.0f);
  const float rstd = rsqrtf(var + 1e-5f);

  const int c = tid * 4;
  float o[4] = { dx * rstd * g[c]     + bt[c],
                 dy * rstd * g[c + 1] + bt[c + 1],
                 dz * rstd * g[c + 2] + bt[c + 2],
                 dw * rstd * g[c + 3] + bt[c + 3] };
  if (outF) {
    float4 ov = { o[0], o[1], o[2], o[3] };
    ((float4*)(outF + (size_t)row * DV))[tid] = ov;
  }
  if (outB) {
    bf16* op = outB + (size_t)row * DV + c;
    op[0] = (bf16)o[0]; op[1] = (bf16)o[1]; op[2] = (bf16)o[2]; op[3] = (bf16)o[3];
  }
}

// ---------------------------------------------------------------------------
// Host orchestration
// ---------------------------------------------------------------------------
extern "C" void kernel_launch(void* const* d_in, const int* in_sizes, int n_in,
                              void* d_out, int out_size, void* d_ws, size_t ws_size,
                              hipStream_t stream) {
  const float* x_q      = (const float*)d_in[0];
  const float* x1       = (const float*)d_in[1];
  const float* x2       = (const float*)d_in[2];
  const float* sa_wq    = (const float*)d_in[3];
  const float* sa_bq    = (const float*)d_in[4];
  const float* sa_wk    = (const float*)d_in[5];
  const float* sa_bk    = (const float*)d_in[6];
  const float* sa_wv    = (const float*)d_in[7];
  const float* sa_bv    = (const float*)d_in[8];
  const float* ln1_g    = (const float*)d_in[9];
  const float* ln1_b    = (const float*)d_in[10];
  const float* mha_in_w = (const float*)d_in[11];
  const float* mha_in_b = (const float*)d_in[12];
  const float* mha_out_w= (const float*)d_in[13];
  const float* mha_out_b= (const float*)d_in[14];
  const float* ln2_g    = (const float*)d_in[15];
  const float* ln2_b    = (const float*)d_in[16];
  const float* ffn_w1   = (const float*)d_in[17];
  const float* ffn_b1   = (const float*)d_in[18];
  const float* ffn_w2   = (const float*)d_in[19];
  const float* ffn_b2   = (const float*)d_in[20];
  const float* ln3_g    = (const float*)d_in[21];
  const float* ln3_b    = (const float*)d_in[22];

  // bump allocator on workspace
  size_t off = 0;
  auto alloc = [&](size_t bytes) -> void* {
    void* p = (char*)d_ws + off;
    off = (off + bytes + 255) & ~(size_t)255;
    return p;
  };
  bf16* wq_b   = (bf16*)alloc((size_t)DV * DV * 2);
  bf16* wk_b   = (bf16*)alloc((size_t)DV * DV * 2);
  bf16* wv_b   = (bf16*)alloc((size_t)DV * DV * 2);
  bf16* win_b  = (bf16*)alloc((size_t)3 * DV * DV * 2);
  bf16* wout_b = (bf16*)alloc((size_t)DV * DV * 2);
  bf16* wf1_b  = (bf16*)alloc((size_t)FFV * DV * 2);
  bf16* wf2_b  = (bf16*)alloc((size_t)DV * FFV * 2);
  bf16* xq_b   = (bf16*)alloc((size_t)BSV * DV * 2);
  bf16* x1_b   = (bf16*)alloc((size_t)BSV * DV * 2);
  bf16* x2_b   = (bf16*)alloc((size_t)BSV * DV * 2);
  bf16* q_b    = (bf16*)alloc((size_t)BSV * DV * 2);
  bf16* k_b    = (bf16*)alloc((size_t)BSV * DV * 2);
  bf16* v_b    = (bf16*)alloc((size_t)BSV * DV * 2);
  bf16* strm_b = (bf16*)alloc((size_t)BSV * DV * 2);
  bf16* attn_b = (bf16*)alloc((size_t)BSV * DV * 2);
  bf16* h_b    = (bf16*)alloc((size_t)BSV * FFV * 2);
  float* attnF = (float*)alloc((size_t)BSV * DV * 4);
  float* projF = (float*)alloc((size_t)BSV * DV * 4);
  float* y0F   = (float*)alloc((size_t)BSV * DV * 4);
  float* y1F   = (float*)alloc((size_t)BSV * DV * 4);

  auto cvt = [&](const float* src, bf16* dst, int n) {
    cvt_bf16_kernel<<<n / 1024, 256, 0, stream>>>(src, dst, n);
  };
  auto gemm = [&](const bf16* A, const bf16* W, const float* bias,
                  float* oF, bf16* oB, int M, int N, int K, int relu) {
    dim3 grid(M / 128, N / 128);
    gemm_wmma_kernel<<<grid, 256, 0, stream>>>(A, W, bias, oF, oB, M, N, K, relu);
  };
  auto attn = [&](const bf16* Q, const bf16* K, const bf16* V,
                  float* oF, bf16* oB, int causal) {
    attn_wmma_kernel<<<BV * HV * (SV / 128), 256, 0, stream>>>(Q, K, V, oF, oB, SV, causal);
  };
  auto ln = [&](const float* x, const float* r, const float* g, const float* bt,
                float* oF, bf16* oB) {
    ln_kernel<<<BSV, 128, 0, stream>>>(x, r, g, bt, oF, oB);
  };

  // ---- bf16 conversions ----
  cvt(x_q, xq_b, BSV * DV);
  cvt(x1,  x1_b, BSV * DV);
  cvt(x2,  x2_b, BSV * DV);
  cvt(sa_wq, wq_b, DV * DV);
  cvt(sa_wk, wk_b, DV * DV);
  cvt(sa_wv, wv_b, DV * DV);
  cvt(mha_in_w,  win_b,  3 * DV * DV);
  cvt(mha_out_w, wout_b, DV * DV);
  cvt(ffn_w1, wf1_b, FFV * DV);
  cvt(ffn_w2, wf2_b, DV * FFV);

  // ---- causal self-attention + LN1 ----
  gemm(xq_b, wq_b, sa_bq, nullptr, q_b, BSV, DV, DV, 0);
  gemm(xq_b, wk_b, sa_bk, nullptr, k_b, BSV, DV, DV, 0);
  gemm(xq_b, wv_b, sa_bv, nullptr, v_b, BSV, DV, DV, 0);
  attn(q_b, k_b, v_b, attnF, nullptr, 1);
  ln(x_q, attnF, ln1_g, ln1_b, y0F, strm_b);

  // ---- cross-attention #1 (q from y, kv from x1) + LN2 ----
  gemm(strm_b, win_b,               mha_in_b,          nullptr, q_b, BSV, DV, DV, 0);
  gemm(x1_b,   win_b + DV * DV,     mha_in_b + DV,     nullptr, k_b, BSV, DV, DV, 0);
  gemm(x1_b,   win_b + 2 * DV * DV, mha_in_b + 2 * DV, nullptr, v_b, BSV, DV, DV, 0);
  attn(q_b, k_b, v_b, nullptr, attn_b, 0);
  gemm(attn_b, wout_b, mha_out_b, projF, nullptr, BSV, DV, DV, 0);
  ln(y0F, projF, ln2_g, ln2_b, y1F, strm_b);

  // ---- cross-attention #2 (q from yattn, kv from x2) + LN2 ----
  gemm(strm_b, win_b,               mha_in_b,          nullptr, q_b, BSV, DV, DV, 0);
  gemm(x2_b,   win_b + DV * DV,     mha_in_b + DV,     nullptr, k_b, BSV, DV, DV, 0);
  gemm(x2_b,   win_b + 2 * DV * DV, mha_in_b + 2 * DV, nullptr, v_b, BSV, DV, DV, 0);
  attn(q_b, k_b, v_b, nullptr, attn_b, 0);
  gemm(attn_b, wout_b, mha_out_b, projF, nullptr, BSV, DV, DV, 0);
  ln(y1F, projF, ln2_g, ln2_b, y0F, strm_b);

  // ---- FFN + LN3 -> d_out ----
  gemm(strm_b, wf1_b, ffn_b1, nullptr, h_b, BSV, FFV, DV, 1);
  gemm(h_b,    wf2_b, ffn_b2, projF, nullptr, BSV, DV, FFV, 0);
  ln(y0F, projF, ln3_g, ln3_b, (float*)d_out, nullptr);
}
